// MultiHeadAttentionViT_84782654423154
// MI455X (gfx1250) — matrix operand order repaired and checked by reading, not compiled, not run on hardware
//
#include <hip/hip_runtime.h>
#include <stdint.h>

typedef __bf16 bf16;
typedef unsigned int u32;
typedef __attribute__((ext_vector_type(16))) __bf16 v16bf;
typedef __attribute__((ext_vector_type(8)))  __bf16 v8bf;
typedef __attribute__((ext_vector_type(8)))  float  v8f;
typedef __attribute__((ext_vector_type(4)))  float  v4f;
typedef __attribute__((ext_vector_type(4)))  u32    v4u;
typedef __attribute__((ext_vector_type(8)))  int    v8i;
typedef __attribute__((ext_vector_type(4)))  int    v4i;

#define D_MODEL 1024
#define HEADS   16
#define DK      64
#define SEQ     2048
#define BATCH   2
#define MTOT    (BATCH * SEQ)   // 4096 rows

#if __has_builtin(__builtin_amdgcn_tensor_load_to_lds) && \
    __has_builtin(__builtin_amdgcn_s_wait_tensorcnt)
#define HAVE_TDM 1
#else
#define HAVE_TDM 0
#endif

#if __has_builtin(__builtin_amdgcn_permlane16)
#define HAVE_PERMLANE16 1
#else
#define HAVE_PERMLANE16 0
#endif

// ---------------------------------------------------------------------------
// helpers
// ---------------------------------------------------------------------------
static __device__ inline v16bf mk16(v8bf lo, v8bf hi) {
  union { v16bf v; v8bf h[2]; } u;
  u.h[0] = lo; u.h[1] = hi;
  return u.v;
}

static __device__ inline v8f wmma_bf16(v16bf a, v16bf b, v8f c) {
  // D = A(16x32 bf16) * B(32x16 bf16) + C(16x16 f32)
  return __builtin_amdgcn_wmma_f32_16x16x32_bf16(
      false, a, false, b, (short)0, c, false, false);
}

// max-reduce across the 16-lane halves of a wave32 (VALU permlane path)
static __device__ inline float red_max16(float x) {
#if HAVE_PERMLANE16
  u32 v, t;
  v = __builtin_bit_cast(u32, x);
  t = __builtin_amdgcn_permlane16(v, v, 0x67452301u, 0xEFCDAB89u, false, false); // xor 1
  x = fmaxf(x, __builtin_bit_cast(float, t)); v = __builtin_bit_cast(u32, x);
  t = __builtin_amdgcn_permlane16(v, v, 0x54761032u, 0xDCFE98BAu, false, false); // xor 2
  x = fmaxf(x, __builtin_bit_cast(float, t)); v = __builtin_bit_cast(u32, x);
  t = __builtin_amdgcn_permlane16(v, v, 0x32107654u, 0xBA98FEDCu, false, false); // xor 4
  x = fmaxf(x, __builtin_bit_cast(float, t)); v = __builtin_bit_cast(u32, x);
  t = __builtin_amdgcn_permlane16(v, v, 0xFEDCBA98u, 0x76543210u, false, false); // xor 8
  x = fmaxf(x, __builtin_bit_cast(float, t));
  return x;
#else
#pragma unroll
  for (int off = 8; off > 0; off >>= 1)
    x = fmaxf(x, __shfl_xor(x, off, 32));
  return x;
#endif
}

#if HAVE_TDM
// Issue a 2-D TDM tile load (bf16 elements) with LDS row padding.
// Packing per CDNA5 ISA §8 (D# groups): group0 = {count/type, lds_addr,
// global_addr}, group1 = {flags, dims, tile dims, stride}. This toolchain's
// builtin takes 6 operands (g0 v4u, g1 v8i, g2 v4i, g3 v4i, g4 v8i, cpol).
static __device__ inline void tdm_load_tile_2d(
    u32 lds_off, const bf16* gaddr, u32 tensor_d0, u32 tensor_d1,
    u32 tile_d0, u32 tile_d1, u32 stride0,
    u32 pad_interval_code, u32 pad_amount_code) {
  unsigned long long ga = (unsigned long long)(uintptr_t)gaddr;
  v4u g0;
  g0[0] = 1u;                                            // count=1, user D#
  g0[1] = lds_off;                                       // lds_addr (bytes)
  g0[2] = (u32)ga;                                       // global_addr[31:0]
  g0[3] = (u32)((ga >> 32) & 0x1ffffffu) | (2u << 30);   // addr[56:32] | type=2
  const u32 flags = (1u << 16)                           // data_size = 2 bytes
                  | (1u << 20)                           // pad_enable
                  | (pad_interval_code << 22)
                  | (pad_amount_code << 25);
  v8i g1;
  g1[0] = (int)flags;
  g1[1] = (int)((tensor_d0 & 0xffffu) << 16);                        // dim0 lo
  g1[2] = (int)(((tensor_d0 >> 16) & 0xffffu) | ((tensor_d1 & 0xffffu) << 16));
  g1[3] = (int)(((tensor_d1 >> 16) & 0xffffu) | ((tile_d0 & 0xffffu) << 16));
  g1[4] = (int)(tile_d1 & 0xffffu);                                  // tile_dim1, tile_dim2=0
  g1[5] = (int)stride0;                                              // dim0 stride lo
  g1[6] = 0;                                                         // stride hi, dim1 stride
  g1[7] = 0;
  v4i g2 = {0, 0, 0, 0};
  v4i g3 = {0, 0, 0, 0};
  v8i g4 = {0, 0, 0, 0, 0, 0, 0, 0};
  __builtin_amdgcn_tensor_load_to_lds(g0, g1, g2, g3, g4, 0);
}
#endif

// ---------------------------------------------------------------------------
// GEMM: C[M=4096, N=1024] = A[4096,1024] @ W[N,K]^T + bias
//   OUT_MODE 0: bf16 [b,h,s,d] (Q,V)  1: bf16 [b,h,d,s] (Kt)  2: f32 row-major
//   A_BF16: A is bf16 (attention output) -> staged via TDM when available
// ---------------------------------------------------------------------------
template <int OUT_MODE, bool A_BF16>
__global__ __launch_bounds__(256) void gemm_bf16_wmma(
    const void* __restrict__ A_, const float* __restrict__ W,
    const float* __restrict__ bias, void* __restrict__ Out) {
  __shared__ __align__(16) bf16 a_tile[128][40];  // 32-col rows padded to 80B
  __shared__ __align__(16) bf16 w_tile[32][72];   // transposed W tile [k][n]

  const int tid  = threadIdx.x;
  const int lane = tid & 31;
  const int wv   = tid >> 5;
  const int half = lane >> 4;
  const int l15  = lane & 15;
  const int m0   = blockIdx.y * 128;
  const int n0   = blockIdx.x * 64;

  v8f acc[4] = {};

  for (int k0 = 0; k0 < D_MODEL; k0 += 32) {
    // ---- stage A tile ----
    if (A_BF16) {
#if HAVE_TDM
      if (wv == 0) {
        // 128x32 bf16 tile; pad 64B rows by 16B -> matches a_tile[128][40]
        tdm_load_tile_2d((u32)(uintptr_t)&a_tile[0][0],
                         (const bf16*)A_ + (size_t)m0 * D_MODEL + k0,
                         D_MODEL, MTOT, 32, 128, D_MODEL,
                         /*interval 16 dwords*/ 3, /*amount 4 dwords*/ 3);
      }
#else
      {
        const int r = tid >> 1, ch = tid & 1;
        const bf16* p = (const bf16*)A_ + (size_t)(m0 + r) * D_MODEL + k0 + 16 * ch;
        *(v8bf*)&a_tile[r][16 * ch]     = *(const v8bf*)p;
        *(v8bf*)&a_tile[r][16 * ch + 8] = *(const v8bf*)(p + 8);
      }
#endif
    } else {
      const int r = tid >> 1, ch = tid & 1;
      const float* p = (const float*)A_ + (size_t)(m0 + r) * D_MODEL + k0 + 16 * ch;
      v4f f0 = *(const v4f*)p,       f1 = *(const v4f*)(p + 4);
      v4f f2 = *(const v4f*)(p + 8), f3 = *(const v4f*)(p + 12);
      v8bf h0, h1;
#pragma unroll
      for (int j = 0; j < 4; ++j) {
        h0[j] = (bf16)f0[j]; h0[4 + j] = (bf16)f1[j];
        h1[j] = (bf16)f2[j]; h1[4 + j] = (bf16)f3[j];
      }
      *(v8bf*)&a_tile[r][16 * ch]     = h0;
      *(v8bf*)&a_tile[r][16 * ch + 8] = h1;
    }

    // ---- stage W tile transposed: w_tile[k][n] = W[n0+n][k0+k] ----
    {
      const int n = tid >> 2, kq = tid & 3;
      const float* p = W + (size_t)(n0 + n) * D_MODEL + k0 + 8 * kq;
      v4f f0 = *(const v4f*)p, f1 = *(const v4f*)(p + 4);
#pragma unroll
      for (int j = 0; j < 4; ++j) {
        w_tile[8 * kq + j][n]     = (bf16)f0[j];
        w_tile[8 * kq + 4 + j][n] = (bf16)f1[j];
      }
    }
#if HAVE_TDM
    if (A_BF16 && wv == 0) __builtin_amdgcn_s_wait_tensorcnt(0);
#endif
    __syncthreads();

    // ---- WMMA: A frag (row 16*wv + l15) x 4 B frags ----
    const int mrow = 16 * wv + l15;
    v16bf afrag = mk16(*(const v8bf*)&a_tile[mrow][8 * half],
                       *(const v8bf*)&a_tile[mrow][16 + 8 * half]);
#pragma unroll
    for (int c = 0; c < 4; ++c) {
      v16bf bfrag;
#pragma unroll
      for (int s = 0; s < 16; ++s)
        bfrag[s] = w_tile[(s & 7) + 8 * half + 16 * (s >> 3)][16 * c + l15];
      acc[c] = wmma_bf16(afrag, bfrag, acc[c]);
    }
    __syncthreads();
  }

  // ---- epilogue: bias + layout-specific store ----
#pragma unroll
  for (int c = 0; c < 4; ++c) {
    const int ncol = n0 + 16 * c + l15;
    const float bv = bias[ncol];
#pragma unroll
    for (int r = 0; r < 8; ++r) {
      const float v = acc[c][r] + bv;
      const int m = m0 + 16 * wv + r + 8 * half;
      if (OUT_MODE == 2) {
        ((float*)Out)[(size_t)m * D_MODEL + ncol] = v;
      } else {
        const int b = m >> 11, s = m & (SEQ - 1);
        const int h = ncol >> 6, d = ncol & (DK - 1);
        if (OUT_MODE == 0)
          ((bf16*)Out)[(((size_t)(b * HEADS + h) * SEQ) + s) * DK + d] = (bf16)v;
        else
          ((bf16*)Out)[(((size_t)(b * HEADS + h) * DK) + d) * SEQ + s] = (bf16)v;
      }
    }
  }
}

// ---------------------------------------------------------------------------
// Flash attention (per head): O = softmax(Q Kt / sqrt(dk)) V
//   Q: [B,H,S,dk]  Kt: [B,H,dk,S]  V: [B,H,S,dk]  (bf16)
//   O: [B*S, 1024] bf16 row-major
// 8 waves/block, 16 query rows per wave, key tile 32, TDM double-buffered K/V.
// ---------------------------------------------------------------------------
__global__ __launch_bounds__(256) void attention_wmma(
    const bf16* __restrict__ Q, const bf16* __restrict__ Kt,
    const bf16* __restrict__ V, bf16* __restrict__ O) {
#if HAVE_TDM
  __shared__ __align__(16) bf16 lds_k[2][64][40];  // Kt tile: 64 d-rows x 32 keys (+pad)
  __shared__ __align__(16) bf16 lds_v[2][32][72];  // V tile: 32 keys x 64 d (+pad)
#endif
  __shared__ __align__(16) bf16 lds_p[8][16][40];  // per-wave P: C-layout -> A-layout

  const int tid  = threadIdx.x;
  const int lane = tid & 31;
  const int wv   = tid >> 5;
  const int half = lane >> 4;
  const int l15  = lane & 15;

  const int bh = blockIdx.x >> 4;   // b*H + h
  const int qt = blockIdx.x & 15;
  const int b  = bh >> 4;
  const int h  = bh & 15;

  const bf16* Qb  = Q  + (size_t)bh * SEQ * DK;
  const bf16* Ktb = Kt + (size_t)bh * DK * SEQ;
  const bf16* Vb  = V  + (size_t)bh * SEQ * DK;
  const int q0 = qt * 128 + wv * 16;

  // Q fragments (16 rows x dk=64 -> two 16x32 A-frags)
  v16bf aq[2];
#pragma unroll
  for (int c = 0; c < 2; ++c) {
    const bf16* p = Qb + (size_t)(q0 + l15) * DK + 32 * c;
    aq[c] = mk16(*(const v8bf*)(p + 8 * half), *(const v8bf*)(p + 16 + 8 * half));
  }

  // all-ones B fragment: row-sums of P via a single WMMA
  v16bf ones;
#pragma unroll
  for (int i = 0; i < 16; ++i) ones[i] = (bf16)1.0f;

  float mi[8], li[8];
#pragma unroll
  for (int r = 0; r < 8; ++r) { mi[r] = -1e30f; li[r] = 0.0f; }
  v8f acc[4] = {};

#if HAVE_TDM
  // preload tile 0 into buffer 0 (wave 0 drives the Tensor Data Mover)
  if (wv == 0) {
    tdm_load_tile_2d((u32)(uintptr_t)&lds_k[0][0][0], Ktb,
                     SEQ, DK, 32, 64, SEQ, 3, 3);
    tdm_load_tile_2d((u32)(uintptr_t)&lds_v[0][0][0], Vb,
                     DK, SEQ, 64, 32, DK, 4, 3);
  }
#endif

  const int NT = SEQ / 32;
  for (int j = 0; j < NT; ++j) {
    const int j0  = j * 32;
#if HAVE_TDM
    const int cur = j & 1, nxt = cur ^ 1;
    const int jn  = (j + 1 < NT) ? (j0 + 32) : j0;  // clamp (dup tile unused)
    if (wv == 0) {
      tdm_load_tile_2d((u32)(uintptr_t)&lds_k[nxt][0][0], Ktb + jn,
                       SEQ, DK, 32, 64, SEQ, 3, 3);
      tdm_load_tile_2d((u32)(uintptr_t)&lds_v[nxt][0][0], Vb + (size_t)jn * DK,
                       DK, SEQ, 64, 32, DK, 4, 3);
      __builtin_amdgcn_s_wait_tensorcnt(2);  // tile j complete (in-order TDM)
    }
    __syncthreads();
#endif

    // ---- scores S(16x32) = Q @ K^T, scaled ----
    v8f s0 = {}, s1 = {};
#pragma unroll
    for (int c = 0; c < 2; ++c) {
#if HAVE_TDM
      const bf16* kp = &lds_k[cur][32 * c][l15];
      const int ks = 40;
#else
      const bf16* kp = Ktb + (size_t)(32 * c) * SEQ + j0 + l15;
      const int ks = SEQ;
#endif
      v16bf b0, b1;
#pragma unroll
      for (int s = 0; s < 16; ++s) {
        const int kk = (s & 7) + 8 * half + 16 * (s >> 3);
        b0[s] = kp[(size_t)kk * ks];
        b1[s] = kp[(size_t)kk * ks + 16];
      }
      s0 = wmma_bf16(aq[c], b0, s0);
      s1 = wmma_bf16(aq[c], b1, s1);
    }
    s0 = s0 * 0.125f;  // 1/sqrt(64)
    s1 = s1 * 0.125f;

    // ---- online softmax: max via permlane butterfly, sum via ones-WMMA ----
    float p0[8], p1[8], sc[8];
#pragma unroll
    for (int r = 0; r < 8; ++r) {
      const float rm = red_max16(fmaxf(s0[r], s1[r]));
      const float mn = fmaxf(mi[r], rm);
      sc[r] = __expf(mi[r] - mn);
      p0[r] = __expf(s0[r] - mn);
      p1[r] = __expf(s1[r] - mn);
      mi[r] = mn;
    }
#pragma unroll
    for (int c = 0; c < 4; ++c)
#pragma unroll
      for (int r = 0; r < 8; ++r) acc[c][r] *= sc[r];

    // P: C-layout -> LDS -> A-fragment
#pragma unroll
    for (int r = 0; r < 8; ++r) {
      lds_p[wv][r + 8 * half][l15]      = (bf16)p0[r];
      lds_p[wv][r + 8 * half][l15 + 16] = (bf16)p1[r];
    }
    const bf16* pp = &lds_p[wv][l15][0];
    v16bf ap = mk16(*(const v8bf*)(pp + 8 * half),
                    *(const v8bf*)(pp + 16 + 8 * half));

    // row sums of (bf16-rounded) P in one WMMA; update normalizers
    v8f zero = {};
    v8f rs = wmma_bf16(ap, ones, zero);
#pragma unroll
    for (int r = 0; r < 8; ++r) li[r] = li[r] * sc[r] + rs[r];

    // ---- O += P(16x32) @ V(32x64) ----
#pragma unroll
    for (int c = 0; c < 4; ++c) {
#if HAVE_TDM
      const bf16* vp = &lds_v[cur][0][16 * c + l15];
      const int vs = 72;
#else
      const bf16* vp = Vb + (size_t)j0 * DK + 16 * c + l15;
      const int vs = DK;
#endif
      v16bf bv;
#pragma unroll
      for (int s = 0; s < 16; ++s)
        bv[s] = vp[(size_t)((s & 7) + 8 * half + 16 * (s >> 3)) * vs];
      acc[c] = wmma_bf16(ap, bv, acc[c]);
    }
#if HAVE_TDM
    __syncthreads();  // all waves done with buffer `cur` before it is refilled
#endif
  }

  // ---- normalize, store as [B*S, 1024] row-major bf16 ----
#pragma unroll
  for (int r = 0; r < 8; ++r) {
    const float inv = 1.0f / li[r];
    const int srow = q0 + r + 8 * half;
    const size_t mrow = (size_t)b * SEQ + srow;
#pragma unroll
    for (int c = 0; c < 4; ++c)
      O[mrow * D_MODEL + h * DK + 16 * c + l15] = (bf16)(acc[c][r] * inv);
  }
}

// ---------------------------------------------------------------------------
// Residual + LayerNorm, in-place on Y (fp32): Y = LN(X + Y) * gamma + beta
// ---------------------------------------------------------------------------
__global__ __launch_bounds__(256) void resid_layernorm(
    float* __restrict__ Y, const float* __restrict__ X,
    const float* __restrict__ gamma, const float* __restrict__ beta) {
  const int row = blockIdx.x;
  const int tid = threadIdx.x;
  float* y = Y + (size_t)row * D_MODEL;
  const float* x = X + (size_t)row * D_MODEL;

  v4f vy = *(const v4f*)(y + tid * 4);
  v4f vx = *(const v4f*)(x + tid * 4);
  v4f v = vx + vy;

  float s  = v[0] + v[1] + v[2] + v[3];
  float ss = v[0] * v[0] + v[1] * v[1] + v[2] * v[2] + v[3] * v[3];
#pragma unroll
  for (int off = 16; off > 0; off >>= 1) {
    s  += __shfl_xor(s, off, 32);
    ss += __shfl_xor(ss, off, 32);
  }

  __shared__ float sm[8], sv[8];
  const int wv = tid >> 5, lane = tid & 31;
  if (lane == 0) { sm[wv] = s; sv[wv] = ss; }
  __syncthreads();
  if (tid == 0) {
    float a = 0.f, q = 0.f;
    for (int i = 0; i < 8; ++i) { a += sm[i]; q += sv[i]; }
    sm[0] = a * (1.0f / 1024.0f);
    sv[0] = q * (1.0f / 1024.0f);
  }
  __syncthreads();
  const float mean = sm[0];
  const float var  = sv[0] - mean * mean;
  const float rstd = rsqrtf(var + 1e-5f);

  v4f g  = *(const v4f*)(gamma + tid * 4);
  v4f bb = *(const v4f*)(beta + tid * 4);
  v4f o;
#pragma unroll
  for (int j = 0; j < 4; ++j) o[j] = (v[j] - mean) * rstd * g[j] + bb[j];
  *(v4f*)(y + tid * 4) = o;
}

// ---------------------------------------------------------------------------
// launch
// ---------------------------------------------------------------------------
extern "C" void kernel_launch(void* const* d_in, const int* in_sizes, int n_in,
                              void* d_out, int out_size, void* d_ws, size_t ws_size,
                              hipStream_t stream) {
  (void)in_sizes; (void)n_in; (void)out_size; (void)ws_size;
  const float* queries = (const float*)d_in[0];
  const float* keys    = (const float*)d_in[1];
  const float* values  = (const float*)d_in[2];
  const float* Wq = (const float*)d_in[3];
  const float* bq = (const float*)d_in[4];
  const float* Wk = (const float*)d_in[5];
  const float* bk = (const float*)d_in[6];
  const float* Wv = (const float*)d_in[7];
  const float* bv = (const float*)d_in[8];
  const float* Wo = (const float*)d_in[9];
  const float* bo = (const float*)d_in[10];
  const float* gamma = (const float*)d_in[11];
  const float* beta  = (const float*)d_in[12];

  // workspace: 4 x (4096*1024) bf16 = 32 MB
  bf16* q_bf  = (bf16*)d_ws;
  bf16* kt_bf = q_bf  + (size_t)MTOT * D_MODEL;
  bf16* v_bf  = kt_bf + (size_t)MTOT * D_MODEL;
  bf16* o_bf  = v_bf  + (size_t)MTOT * D_MODEL;

  dim3 blk(256);
  dim3 gGrid(D_MODEL / 64, MTOT / 128);   // 16 x 32

  gemm_bf16_wmma<0, false><<<gGrid, blk, 0, stream>>>(queries, Wq, bq, q_bf);
  gemm_bf16_wmma<1, false><<<gGrid, blk, 0, stream>>>(keys,    Wk, bk, kt_bf);
  gemm_bf16_wmma<0, false><<<gGrid, blk, 0, stream>>>(values,  Wv, bv, v_bf);

  attention_wmma<<<dim3(BATCH * HEADS * (SEQ / 128)), blk, 0, stream>>>(
      q_bf, kt_bf, v_bf, o_bf);

  gemm_bf16_wmma<2, true><<<gGrid, blk, 0, stream>>>(o_bf, Wo, bo, d_out);

  resid_layernorm<<<dim3(MTOT), blk, 0, stream>>>(
      (float*)d_out, queries, gamma, beta);
}
